// TransformerEncoder_4535485464717
// MI455X (gfx1250) — compile-verified, matrix operand order checked by reading, not run on hardware
//
#include <hip/hip_runtime.h>
#include <hip/hip_bf16.h>

// ---------------------------------------------------------------------------
// Transformer encoder forward for MI455X (gfx1250).
// All GEMMs go through v_wmma_f32_16x16x32_bf16 (bf16 A/B, fp32 accumulate).
// Each wave computes a 16x64 strip (4 accumulators) to amortize A fragments.
// ---------------------------------------------------------------------------

#define Bc   4
#define Tc   1024
#define Dc   1024
#define Hc   16
#define DKc  64
#define DFFc 4096
#define NLAY 3

typedef __attribute__((ext_vector_type(16))) __bf16 v16bf;
typedef __attribute__((ext_vector_type(8)))  __bf16 v8bf;
typedef __attribute__((ext_vector_type(8)))  float  v8f;
typedef __attribute__((ext_vector_type(4)))  float  v4f;

// ---------------------------------------------------------------------------
// Embedding + sinusoidal positional encoding:  x = embed[src]*sqrt(D) + PE
// ---------------------------------------------------------------------------
__global__ void k_embed(const int* __restrict__ src, const float* __restrict__ embed,
                        float* __restrict__ x)
{
    size_t i = (size_t)blockIdx.x * 256 + threadIdx.x;          // < B*T*D
    int d   = (int)(i % Dc);
    size_t bt = i / Dc;
    int t   = (int)(bt % Tc);
    int tok = src[bt];
    float div = __expf((float)(d & ~1) * (-9.210340371976184f / (float)Dc));
    float ang = (float)t * div;
    float pe  = (d & 1) ? cosf(ang) : sinf(ang);
    x[i] = embed[(size_t)tok * Dc + d] * 32.0f + pe;            // sqrt(1024)=32
}

// ---------------------------------------------------------------------------
// Weight prep: W fp32 [K,N] -> Wt bf16 [N,K] (transpose + convert).
// ---------------------------------------------------------------------------
__global__ void k_w_to_bf16t(const float* __restrict__ W, __bf16* __restrict__ Wt,
                             int Kd, int Nd)
{
    size_t i = (size_t)blockIdx.x * 256 + threadIdx.x;          // < Nd*Kd
    if (i >= (size_t)Nd * Kd) return;
    int n = (int)(i / Kd);
    int k = (int)(i % Kd);
    Wt[i] = (__bf16)W[(size_t)k * Nd + n];
}

// fp32 -> bf16 elementwise (same layout); used for the K matrix.
__global__ void k_f32_to_bf16(const float* __restrict__ in, __bf16* __restrict__ out,
                              size_t n)
{
    size_t i = (size_t)blockIdx.x * 256 + threadIdx.x;
    if (i < n) out[i] = (__bf16)in[i];
}

// V [b,t,h,dk] fp32  ->  Vt [(b*H+h)*DK + dk][T] bf16  (K-contiguous operand)
__global__ void k_v_transpose(const float* __restrict__ V, __bf16* __restrict__ Vt)
{
    size_t i = (size_t)blockIdx.x * 256 + threadIdx.x;          // < B*H*DK*T
    int t  = (int)(i % Tc);
    size_t r = i / Tc;
    int dk = (int)(r % DKc); r /= DKc;
    int h  = (int)(r % Hc);
    int b  = (int)(r / Hc);
    Vt[i] = (__bf16)V[(size_t)(b * Tc + t) * Dc + h * DKc + dk];
}

// ---------------------------------------------------------------------------
// Fragment loaders (16-bit WMMA operand layout, wave32):
//  lanes 0-15 hold K = 0..7 / 16..23, lanes 16-31 hold K = 8..15 / 24..31.
// ---------------------------------------------------------------------------
__device__ __forceinline__ v16bf load_afrag_f32(const float* p)
{
    v4f f0 = *(const v4f*)(p);
    v4f f1 = *(const v4f*)(p + 4);
    v4f f2 = *(const v4f*)(p + 16);
    v4f f3 = *(const v4f*)(p + 20);
    v16bf a;
#pragma unroll
    for (int j = 0; j < 4; ++j) {
        a[j]      = (__bf16)f0[j];
        a[4 + j]  = (__bf16)f1[j];
        a[8 + j]  = (__bf16)f2[j];
        a[12 + j] = (__bf16)f3[j];
    }
    return a;
}

__device__ __forceinline__ v16bf load_bfrag_bf16(const __bf16* p)
{
    v8bf b0 = *(const v8bf*)(p);
    v8bf b1 = *(const v8bf*)(p + 16);
    v16bf b;
#pragma unroll
    for (int j = 0; j < 8; ++j) { b[j] = b0[j]; b[8 + j] = b1[j]; }
    return b;
}

// ---------------------------------------------------------------------------
// Batched WMMA GEMM:   C[z] = act( A[z](fp32) * Bt[z](bf16,[N,K]) * scale + bias )
// Block: 256 thr = 8 waves stacked in M; block tile 128(M) x 64(N).
// Wave tile: 16(M) x 64(N) = 4 accumulators sharing one A fragment.
// z decomposes as (zb = z/Hz, zh = z%Hz) with independent strides for A/B/C.
// ---------------------------------------------------------------------------
__global__ void __launch_bounds__(256)
k_gemm(const float* __restrict__ A, const __bf16* __restrict__ Bt,
       const float* __restrict__ bias, float* __restrict__ C,
       int M, int Nn, int K, int lda, int ldb, int ldc,
       float scale, int relu, int Hz,
       long long aZb, long long aZh, long long bZb, long long bZh,
       long long cZb, long long cZh)
{
    int z  = blockIdx.z;
    int zb = z / Hz, zh = z % Hz;
    const float*  Az = A  + zb * aZb + zh * aZh;
    const __bf16* Bz = Bt + zb * bZb + zh * bZh;
    float*        Cz = C  + zb * cZb + zh * cZh;

    int w    = threadIdx.x >> 5;
    int lane = threadIdx.x & 31;
    int m0   = blockIdx.y * 128 + w * 16;
    int n0   = blockIdx.x * 64;
    if (m0 >= M || n0 >= Nn) return;

    int r  = lane & 15;          // row-in-tile (A) / col-in-tile (B,C)
    int hf = lane >> 4;          // half-select: K sub-block / M sub-block

    size_t aRow = (size_t)(m0 + r) * lda;
    size_t bRow = (size_t)(n0 + r) * ldb;
    const long long bTile = (long long)16 * ldb;   // N-tile stride in Bt

    v8f acc[4] = {};

    for (int kk = 0; kk < K; kk += 32) {
        const float* pa = Az + aRow + kk + hf * 8;
        if (kk + 32 < K) __builtin_prefetch(pa + 32, 0, 3);   // next K-chunk of A
        v16bf a = load_afrag_f32(pa);

        const __bf16* pb = Bz + bRow + kk + hf * 8;
#pragma unroll
        for (int nt = 0; nt < 4; ++nt) {
            v16bf b = load_bfrag_bf16(pb + nt * bTile);
            acc[nt] = __builtin_amdgcn_wmma_f32_16x16x32_bf16(
                          false, a, false, b, (short)0, acc[nt], false, false);
        }
    }

    // ---- epilogue: lane%16 = N, VGPR i -> M = i + 8*(lane>=16)
#pragma unroll
    for (int nt = 0; nt < 4; ++nt) {
        int   n  = n0 + nt * 16 + r;
        float bvv = bias ? bias[n] : 0.0f;
#pragma unroll
        for (int i = 0; i < 8; ++i) {
            int   m = m0 + i + hf * 8;
            float v = acc[nt][i] * scale + bvv;
            if (relu) v = fmaxf(v, 0.0f);
            Cz[(size_t)m * ldc + n] = v;
        }
    }
}

// ---------------------------------------------------------------------------
// Row softmax, in place (rows of length T).
// ---------------------------------------------------------------------------
__global__ void k_softmax(float* __restrict__ s)
{
    float* p = s + (size_t)blockIdx.x * Tc;
    int tid = threadIdx.x;
    __shared__ float red[256];

    float m = -1e30f;
    for (int j = tid; j < Tc; j += 256) m = fmaxf(m, p[j]);
    red[tid] = m; __syncthreads();
    for (int st = 128; st > 0; st >>= 1) {
        if (tid < st) red[tid] = fmaxf(red[tid], red[tid + st]);
        __syncthreads();
    }
    m = red[0]; __syncthreads();

    float sum = 0.0f;
    for (int j = tid; j < Tc; j += 256) {
        float e = __expf(p[j] - m);
        p[j] = e; sum += e;
    }
    red[tid] = sum; __syncthreads();
    for (int st = 128; st > 0; st >>= 1) {
        if (tid < st) red[tid] += red[tid + st];
        __syncthreads();
    }
    float inv = 1.0f / red[0];
    for (int j = tid; j < Tc; j += 256) p[j] *= inv;
}

// ---------------------------------------------------------------------------
// out = LayerNorm(x + y) * g + b   (y may be null). One block per row, D=1024.
// ---------------------------------------------------------------------------
__global__ void k_res_ln(const float* __restrict__ xin, const float* __restrict__ yin,
                         const float* __restrict__ g, const float* __restrict__ b,
                         float* __restrict__ out)
{
    size_t row = blockIdx.x;
    int tid = threadIdx.x;
    const float* xr = xin + row * Dc;
    const float* yr = yin ? yin + row * Dc : nullptr;

    float v[4];
    float s = 0.0f, ss = 0.0f;
#pragma unroll
    for (int j = 0; j < 4; ++j) {
        int d = tid + j * 256;
        float t = xr[d] + (yr ? yr[d] : 0.0f);
        v[j] = t; s += t; ss += t * t;
    }
    __shared__ float rs[256], rss[256];
    rs[tid] = s; rss[tid] = ss; __syncthreads();
    for (int st = 128; st > 0; st >>= 1) {
        if (tid < st) { rs[tid] += rs[tid + st]; rss[tid] += rss[tid + st]; }
        __syncthreads();
    }
    float mean = rs[0] * (1.0f / Dc);
    float var  = rss[0] * (1.0f / Dc) - mean * mean;
    float rstd = rsqrtf(var + 1e-5f);
#pragma unroll
    for (int j = 0; j < 4; ++j) {
        int d = tid + j * 256;
        out[row * Dc + d] = (v[j] - mean) * rstd * g[d] + b[d];
    }
}

// ---------------------------------------------------------------------------
// Host-side orchestration (graph-capture safe: only launches on `stream`).
// ---------------------------------------------------------------------------
extern "C" void kernel_launch(void* const* d_in, const int* in_sizes, int n_in,
                              void* d_out, int out_size, void* d_ws, size_t ws_size,
                              hipStream_t stream)
{
    (void)in_sizes; (void)n_in; (void)out_size; (void)ws_size;

    const int*   src   = (const int*)  d_in[0];
    const float* embed = (const float*)d_in[1];
    const float* Wq    = (const float*)d_in[2];
    const float* bq    = (const float*)d_in[3];
    const float* Wk    = (const float*)d_in[4];
    const float* bk    = (const float*)d_in[5];
    const float* Wv    = (const float*)d_in[6];
    const float* bv    = (const float*)d_in[7];
    const float* Wo    = (const float*)d_in[8];
    const float* bo    = (const float*)d_in[9];
    const float* ln1g  = (const float*)d_in[10];
    const float* ln1b  = (const float*)d_in[11];
    const float* W1    = (const float*)d_in[12];
    const float* b1    = (const float*)d_in[13];
    const float* W2    = (const float*)d_in[14];
    const float* b2    = (const float*)d_in[15];
    const float* ln2g  = (const float*)d_in[16];
    const float* ln2b  = (const float*)d_in[17];
    const float* fg    = (const float*)d_in[18];
    const float* fb    = (const float*)d_in[19];

    const size_t MT = (size_t)Bc * Tc;          // 4096 rows
    const size_t DD = (size_t)Dc * Dc;          // 1M
    const size_t DF = (size_t)Dc * DFFc;        // 4M

    // ---- workspace carve-up (~250 MB) ----
    char* wsb = (char*)d_ws;
    auto carve = [&](size_t bytes) -> char* {
        char* p = wsb; wsb += (bytes + 255) & ~(size_t)255; return p;
    };
    float*  x    = (float*) carve(MT * Dc * 4);
    float*  q    = (float*) carve(MT * Dc * 4);
    float*  kf   = (float*) carve(MT * Dc * 4);
    float*  vf   = (float*) carve(MT * Dc * 4);
    float*  ctx  = (float*) carve(MT * Dc * 4);
    float*  proj = (float*) carve(MT * Dc * 4);
    float*  ffnh = (float*) carve(MT * DFFc * 4);
    float*  ffno = (float*) carve(MT * Dc * 4);
    __bf16* kbf  = (__bf16*)carve(MT * Dc * 2);
    __bf16* vt   = (__bf16*)carve((size_t)Bc * Hc * DKc * Tc * 2);
    __bf16* wqt  = (__bf16*)carve(NLAY * DD * 2);
    __bf16* wkt  = (__bf16*)carve(NLAY * DD * 2);
    __bf16* wvt  = (__bf16*)carve(NLAY * DD * 2);
    __bf16* wot  = (__bf16*)carve(NLAY * DD * 2);
    __bf16* w1t  = (__bf16*)carve(NLAY * DF * 2);
    __bf16* w2t  = (__bf16*)carve(NLAY * DF * 2);

    float* x_out    = (float*)d_out;
    float* attn_out = x_out + MT * Dc;

    auto blocks = [](size_t n) { return (unsigned)((n + 255) / 256); };

    // ---- weight prep: transpose + bf16 ----
    for (int i = 0; i < NLAY; ++i) {
        k_w_to_bf16t<<<blocks(DD), 256, 0, stream>>>(Wq + i * DD, wqt + i * DD, Dc, Dc);
        k_w_to_bf16t<<<blocks(DD), 256, 0, stream>>>(Wk + i * DD, wkt + i * DD, Dc, Dc);
        k_w_to_bf16t<<<blocks(DD), 256, 0, stream>>>(Wv + i * DD, wvt + i * DD, Dc, Dc);
        k_w_to_bf16t<<<blocks(DD), 256, 0, stream>>>(Wo + i * DD, wot + i * DD, Dc, Dc);
        k_w_to_bf16t<<<blocks(DF), 256, 0, stream>>>(W1 + i * DF, w1t + i * DF, Dc, DFFc);
        k_w_to_bf16t<<<blocks(DF), 256, 0, stream>>>(W2 + i * DF, w2t + i * DF, DFFc, Dc);
    }

    // ---- embedding + positional encoding ----
    k_embed<<<blocks(MT * Dc), 256, 0, stream>>>(src, embed, x);

    auto gemm = [&](const float* A, const __bf16* Bt, const float* bias, float* C,
                    int M, int Nn, int K, int lda, int ldb, int ldc,
                    float scale, int relu, int Z, int Hz,
                    long long aZb, long long aZh, long long bZb, long long bZh,
                    long long cZb, long long cZh) {
        dim3 g((unsigned)(Nn / 64), (unsigned)(M / 128), (unsigned)Z);
        k_gemm<<<g, 256, 0, stream>>>(A, Bt, bias, C, M, Nn, K, lda, ldb, ldc,
                                      scale, relu, Hz, aZb, aZh, bZb, bZh, cZb, cZh);
    };

    for (int i = 0; i < NLAY; ++i) {
        // Q, K, V projections: [4096,1024] x [1024,1024]
        gemm(x, wqt + i * DD, bq + i * Dc, q,  (int)MT, Dc, Dc, Dc, Dc, Dc, 1.f, 0, 1, 1, 0,0,0,0,0,0);
        gemm(x, wkt + i * DD, bk + i * Dc, kf, (int)MT, Dc, Dc, Dc, Dc, Dc, 1.f, 0, 1, 1, 0,0,0,0,0,0);
        gemm(x, wvt + i * DD, bv + i * Dc, vf, (int)MT, Dc, Dc, Dc, Dc, Dc, 1.f, 0, 1, 1, 0,0,0,0,0,0);

        k_f32_to_bf16<<<blocks(MT * Dc), 256, 0, stream>>>(kf, kbf, MT * Dc);
        k_v_transpose<<<blocks(MT * Dc), 256, 0, stream>>>(vf, vt);

        float* attn_i = attn_out + (size_t)i * Bc * Hc * Tc * Tc;

        // scores = Q K^T / sqrt(DK), written (pre-softmax) straight into d_out slot
        gemm(q, kbf, nullptr, attn_i, Tc, Tc, DKc, Dc, Dc, Tc, 0.125f, 0,
             Bc * Hc, Hc,
             (long long)Tc * Dc, (long long)DKc,          // A: +b*T*D, +h*DK
             (long long)Tc * Dc, (long long)DKc,          // B: same layout as A
             (long long)Hc * Tc * Tc, (long long)Tc * Tc);// C: +z*T*T

        k_softmax<<<(unsigned)(Bc * Hc * Tc), 256, 0, stream>>>(attn_i);

        // context = attn @ V : [1024,1024] x [1024,64] per (b,h)
        gemm(attn_i, vt, nullptr, ctx, Tc, DKc, Tc, Tc, Tc, Dc, 1.f, 0,
             Bc * Hc, Hc,
             (long long)Hc * Tc * Tc, (long long)Tc * Tc,
             (long long)Hc * DKc * Tc, (long long)DKc * Tc,
             (long long)Tc * Dc, (long long)DKc);

        // output projection + residual LN1
        gemm(ctx, wot + i * DD, bo + i * Dc, proj, (int)MT, Dc, Dc, Dc, Dc, Dc, 1.f, 0, 1, 1, 0,0,0,0,0,0);
        k_res_ln<<<(unsigned)MT, 256, 0, stream>>>(x, proj, ln1g + i * Dc, ln1b + i * Dc, x);

        // FFN
        gemm(x, w1t + i * DF, b1 + i * DFFc, ffnh, (int)MT, DFFc, Dc, Dc, Dc, DFFc, 1.f, 1, 1, 1, 0,0,0,0,0,0);
        gemm(ffnh, w2t + i * DF, b2 + i * Dc, ffno, (int)MT, Dc, DFFc, DFFc, DFFc, Dc, 1.f, 0, 1, 1, 0,0,0,0,0,0);
        k_res_ln<<<(unsigned)MT, 256, 0, stream>>>(x, ffno, ln2g + i * Dc, ln2b + i * Dc, x);
    }

    // final LayerNorm into d_out
    k_res_ln<<<(unsigned)MT, 256, 0, stream>>>(x, nullptr, fg, fb, x_out);
}
